// MolNet_46514495816182
// MI455X (gfx1250) — matrix operand order repaired and checked by reading, not compiled, not run on hardware
//
#include <hip/hip_runtime.h>
#include <stdint.h>

typedef uint32_t u32;
typedef __attribute__((ext_vector_type(16))) __bf16 bf16x16;
typedef __attribute__((ext_vector_type(8)))  float  f32x8;
typedef __attribute__((ext_vector_type(4)))  u32    u32x4;

union FragBF { bf16x16 v; u32x4 q[2]; };

__device__ __forceinline__ uint16_t f2b(float f) {
  u32 u = __float_as_uint(f);
  u += 0x7FFFu + ((u >> 16) & 1u);           // round-to-nearest-even
  return (uint16_t)(u >> 16);
}
__device__ __forceinline__ float b2f(uint16_t h) {
  return __uint_as_float(((u32)h) << 16);
}

// A fragment: 16x32 bf16, lane L holds row M=L%16; halves 0..7 = K[kk+8g .. +7],
// halves 8..15 = K[kk+16+8g .. +7]  -> two contiguous 16B loads from row-major A.
__device__ __forceinline__ void loadA(FragBF& f, const uint16_t* p) {
  f.q[0] = *(const u32x4*)p;
  f.q[1] = *(const u32x4*)(p + 16);
}
// B fragment: 32x16 bf16, lane L holds col N=L%16; halves 0..7 = K[kk+8g .. +7],
// halves 8..15 = K[kk+16+8g .. +7]  -> two contiguous 16B loads from row-major B^T (N x K).
__device__ __forceinline__ void loadB(FragBF& f, const uint16_t* p) {
  f.q[0] = *(const u32x4*)p;
  f.q[1] = *(const u32x4*)(p + 16);
}
__device__ __forceinline__ f32x8 wmma_bf16(const FragBF& a, const FragBF& b, f32x8 c) {
  return __builtin_amdgcn_wmma_f32_16x16x32_bf16(false, a.v, false, b.v, (short)0, c, false, false);
}

enum { ACT_NONE = 0, ACT_RELU = 1, ACT_LRELU = 2 };

// ---------------------------------------------------------------------------
// Generic bf16 WMMA GEMM: C[M,N] = act(A[M,K] @ Bt[N,K]^T + bias [+ Cold])
// 32x64 macro-tile per wave: 2 A fragments reused across 4 B fragments per
// k-step -> 8 WMMAs from 6 fragment loads; 4x less A-side L2 traffic than a
// 16x16-per-wave scheme. Requires M%32==0, N%64==0 (true for all call sites).
// ---------------------------------------------------------------------------
template<int ACT, bool OUTB, bool ACC>
__global__ __launch_bounds__(256) void gemm_bf16_kernel(
    const uint16_t* __restrict__ A, const uint16_t* __restrict__ Bt,
    const float* __restrict__ bias, float* __restrict__ Cf, uint16_t* __restrict__ Cb,
    int M, int N, int K)
{
  int wid = (int)((blockIdx.x * 256u + threadIdx.x) >> 5);
  int tilesN = N >> 6;                          // 64-wide column strips
  int total  = (M >> 5) * tilesN;               // 32-high row strips
  if (wid >= total) return;                     // wave-uniform exit (EXEC all-ones)
  int tm = wid / tilesN, tnb = wid - tm * tilesN;
  int lane = threadIdx.x & 31;
  int g = lane >> 4, r = lane & 15;
  long arow0 = (long)(tm * 32 + r) * K;
  long arow1 = arow0 + (long)16 * K;
  long brow[4];
#pragma unroll
  for (int j = 0; j < 4; j++) brow[j] = (long)(tnb * 64 + j * 16 + r) * K;
  f32x8 acc[2][4];
#pragma unroll
  for (int h = 0; h < 2; h++)
#pragma unroll
    for (int j = 0; j < 4; j++) acc[h][j] = (f32x8){};
  for (int kk = 0; kk < K; kk += 32) {
    FragBF fa0, fa1;
    loadA(fa0, A + arow0 + kk + 8 * g);
    loadA(fa1, A + arow1 + kk + 8 * g);
    if (kk + 32 < K) {                          // hint next k-slice into cache
      __builtin_prefetch(A + arow0 + kk + 32, 0, 0);
      __builtin_prefetch(A + arow1 + kk + 32, 0, 0);
    }
#pragma unroll
    for (int j = 0; j < 4; j++) {
      FragBF fb;
      loadB(fb, Bt + brow[j] + kk + 8 * g);
      acc[0][j] = wmma_bf16(fa0, fb, acc[0][j]);
      acc[1][j] = wmma_bf16(fa1, fb, acc[1][j]);
    }
  }
#pragma unroll
  for (int j = 0; j < 4; j++) {
    int n = tnb * 64 + j * 16 + r;
    float bv = bias ? bias[n] : 0.0f;
#pragma unroll
    for (int h = 0; h < 2; h++) {
#pragma unroll
      for (int i = 0; i < 8; i++) {
        int m = tm * 32 + h * 16 + i + 8 * g;   // D layout: VGPR i -> M = i + 8*(lane/16)
        long idx = (long)m * N + n;
        float v = acc[h][j][i] + bv;
        if (ACC) v += Cf[idx];
        if (ACT == ACT_RELU)  v = fmaxf(v, 0.0f);
        if (ACT == ACT_LRELU) v = (v >= 0.0f) ? v : 0.01f * v;
        if (OUTB) Cb[idx] = f2b(v); else Cf[idx] = v;
      }
    }
  }
}

// ---------------------------------------------------------------------------
// Small utility kernels
// ---------------------------------------------------------------------------
__global__ void fill_f32_kernel(float* p, float v, long n) {
  long i = (long)blockIdx.x * 256 + threadIdx.x; if (i < n) p[i] = v;
}
__global__ void fill_u32_kernel(u32* p, u32 v, long n) {
  long i = (long)blockIdx.x * 256 + threadIdx.x; if (i < n) p[i] = v;
}
__global__ void f32_to_bf16_kernel(const float* __restrict__ in, uint16_t* __restrict__ out, long n) {
  long i = (long)blockIdx.x * 256 + threadIdx.x; if (i < n) out[i] = f2b(in[i]);
}
// Wt[n*K+k] = W[k*N+n] (bf16)
__global__ void transpose_to_bf16(const float* __restrict__ W, uint16_t* __restrict__ Wt, int K, int N) {
  long id = (long)blockIdx.x * 256 + threadIdx.x;
  if (id >= (long)K * N) return;
  int n = (int)(id / K), k = (int)(id % K);
  Wt[id] = f2b(W[(long)k * N + n]);
}

// ---------------------------------------------------------------------------
// Phase A: edge geometry, vnin assembly
// ---------------------------------------------------------------------------
__global__ void edge_geom_kernel(const float* __restrict__ c, const int* __restrict__ src,
                                 const int* __restrict__ dst, float* __restrict__ ps2t, int E) {
  int e = blockIdx.x * 256 + threadIdx.x; if (e >= E) return;
  int s0 = src[e], d0 = dst[e];
#pragma unroll
  for (int j = 0; j < 3; j++) ps2t[(long)e * 3 + j] = c[(long)d0 * 3 + j] - c[(long)s0 * 3 + j];
}

__global__ void vnin_kernel(const float* __restrict__ ps2t, const int* __restrict__ src,
                            const int* __restrict__ dst, const float* __restrict__ type_emb,
                            const int* __restrict__ x, const float* __restrict__ posi_W,
                            const float* __restrict__ posi_b, uint16_t* __restrict__ vnin, int E) {
  long id = (long)blockIdx.x * 256 + threadIdx.x;
  if (id >= (long)E * 64) return;
  long e = id >> 6; int cch = (int)(id & 63);
  int ts = x[(long)src[e] * 6];
  int td = x[(long)dst[e] * 6];
  float p0 = ps2t[e * 3], p1 = ps2t[e * 3 + 1], p2 = ps2t[e * 3 + 2];
  float vp = posi_b[cch] + p0 * posi_W[cch] + p1 * posi_W[64 + cch] + p2 * posi_W[128 + cch];
  vnin[e * 192 + cch]       = f2b(type_emb[(long)ts * 64 + cch]);
  vnin[e * 192 + 64 + cch]  = f2b(vp);
  vnin[e * 192 + 128 + cch] = f2b(type_emb[(long)td * 64 + cch]);
}

// ---------------------------------------------------------------------------
// Per-graph geometry kernel: cos = pos pos^T / denom, tmp = cos@v, feat = v^T@tmp
// One workgroup (8 waves) per graph; all WMMA operands in LDS.
// ---------------------------------------------------------------------------
__global__ __launch_bounds__(256) void feat_kernel(const float* __restrict__ ps2t,
                                                   const uint16_t* __restrict__ vn,
                                                   uint16_t* __restrict__ feat) {
  __shared__ __align__(16) uint16_t svT[64 * 128];    // v transposed [d][e]
  __shared__ __align__(16) uint16_t sbuf[128 * 128];  // cos, later tmp^T [d][e]
  __shared__ float spos[384];
  __shared__ float sred[256];
  int gph = blockIdx.x, tid = threadIdx.x;
  for (int idx = tid; idx < 8192; idx += 256) {
    int e = idx >> 6, d = idx & 63;
    svT[d * 128 + e] = vn[(long)gph * 8192 + idx];
  }
  for (int idx = tid; idx < 384; idx += 256) spos[idx] = ps2t[(long)gph * 384 + idx];
  __syncthreads();
  float part = 0.0f;
  for (int idx = tid; idx < 384; idx += 256) part += spos[idx] * spos[idx];
  sred[tid] = part;
  __syncthreads();
  for (int s2 = 128; s2 > 0; s2 >>= 1) {
    if (tid < s2) sred[tid] += sred[tid + s2];
    __syncthreads();
  }
  float inv = 1.0f / sred[0];
  for (int idx = tid; idx < 16384; idx += 256) {
    int i = idx >> 7, j = idx & 127;
    float d3 = spos[i * 3] * spos[j * 3] + spos[i * 3 + 1] * spos[j * 3 + 1] + spos[i * 3 + 2] * spos[j * 3 + 2];
    sbuf[idx] = f2b(d3 * inv);
  }
  __syncthreads();
  int wid = tid >> 5, lane = tid & 31, g = lane >> 4, r = lane & 15;
  // tmp = cos(128x128) @ v(128x64): 32 tiles, 4 per wave, kept in registers
  f32x8 acc[4];
#pragma unroll
  for (int q2 = 0; q2 < 4; q2++) {
    int t = wid * 4 + q2, tm = t >> 2, tn = t & 3;
    f32x8 a = {};
    for (int kk = 0; kk < 128; kk += 32) {
      FragBF fa, fb;
      loadA(fa, sbuf + (tm * 16 + r) * 128 + kk + 8 * g);
      loadB(fb, svT + (tn * 16 + r) * 128 + kk + 8 * g);
      a = wmma_bf16(fa, fb, a);
    }
    acc[q2] = a;
  }
  __syncthreads();   // cos fully consumed
#pragma unroll
  for (int q2 = 0; q2 < 4; q2++) {
    int t = wid * 4 + q2, tm = t >> 2, tn = t & 3;
    int n = tn * 16 + r;
#pragma unroll
    for (int i = 0; i < 8; i++) {
      int m = tm * 16 + i + 8 * g;
      sbuf[n * 128 + m] = f2b(acc[q2][i]);     // store tmp transposed [d][e]
    }
  }
  __syncthreads();
  // feat[d][k] = sum_e v[e][d]*tmp[e][k]: A = svT rows, Bt = tmp^T rows, K=128
  for (int t = wid; t < 16; t += 8) {
    int tm = t >> 2, tn = t & 3;
    f32x8 a = {};
    for (int kk = 0; kk < 128; kk += 32) {
      FragBF fa, fb;
      loadA(fa, svT + (tm * 16 + r) * 128 + kk + 8 * g);
      loadB(fb, sbuf + (tn * 16 + r) * 128 + kk + 8 * g);
      a = wmma_bf16(fa, fb, a);
    }
    int n = tn * 16 + r;
#pragma unroll
    for (int i = 0; i < 8; i++) {
      int m = tm * 16 + i + 8 * g;
      feat[(long)gph * 4096 + m * 64 + n] = f2b(a[i]);
    }
  }
}

__global__ void ec_kernel(const float* __restrict__ h2, const float* __restrict__ W3,
                          const float* __restrict__ b3, float* __restrict__ out, int G) {
  int g = blockIdx.x * 256 + threadIdx.x; if (g >= G) return;
  float s = b3[0];
#pragma unroll
  for (int d = 0; d < 64; d++) s += h2[(long)g * 64 + d] * W3[d];
  out[g] += s;
}

__global__ void ea_energy_kernel(const int* __restrict__ x, const int* __restrict__ batch,
                                 const float* __restrict__ ee, float* __restrict__ out, int N) {
  int n = blockIdx.x * 256 + threadIdx.x; if (n >= N) return;
  atomicAdd(&out[batch[n]], ee[x[(long)n * 6]]);
}

// ---------------------------------------------------------------------------
// Node / edge feature kernels
// ---------------------------------------------------------------------------
__global__ void xf_kernel(const int* __restrict__ x, const float* __restrict__ W,
                          const float* __restrict__ b, float* __restrict__ hf,
                          uint16_t* __restrict__ hb, int N) {
  long id = (long)blockIdx.x * 256 + threadIdx.x;
  if (id >= (long)N * 64) return;
  long n = id >> 6; int cch = (int)(id & 63);
  float s = b[cch];
#pragma unroll
  for (int j = 0; j < 6; j++) s += (float)x[n * 6 + j] * W[j * 64 + cch];
  hf[id] = s; hb[id] = f2b(s);
}

__global__ void ea0_kernel(const int* __restrict__ eattr, const float* __restrict__ emb,
                           uint16_t* __restrict__ out, int E) {
  long id = (long)blockIdx.x * 256 + threadIdx.x;
  if (id >= (long)E * 64) return;
  long e = id >> 6; int cch = (int)(id & 63);
  out[id] = f2b(emb[(long)eattr[e] * 64 + cch]);
}

__global__ void cnt_kernel(const int* __restrict__ dst, float* __restrict__ cnt, int E) {
  int e = blockIdx.x * 256 + threadIdx.x; if (e >= E) return;
  atomicAdd(&cnt[dst[e]], 1.0f);
}
__global__ void loop_acc_kernel(const int* __restrict__ dst, const float* __restrict__ ea,
                                float* __restrict__ loopb, int E) {
  long id = (long)blockIdx.x * 256 + threadIdx.x;
  if (id >= (long)E * 64) return;
  long e = id >> 6; int cch = (int)(id & 63);
  atomicAdd(&loopb[(long)dst[e] * 64 + cch], ea[id]);
}
__global__ void loop_fin_kernel(const float* __restrict__ loopb, const float* __restrict__ cnt,
                                uint16_t* __restrict__ out, int N) {
  long id = (long)blockIdx.x * 256 + threadIdx.x;
  if (id >= (long)N * 64) return;
  long n = id >> 6;
  out[id] = f2b(loopb[id] / fmaxf(cnt[n], 1.0f));
}

// ---------------------------------------------------------------------------
// GAT attention kernels
// ---------------------------------------------------------------------------
__global__ void ah_kernel(const float* __restrict__ hh, const float* __restrict__ asrc,
                          const float* __restrict__ adst, float* __restrict__ ah_s,
                          float* __restrict__ ah_d, int N) {
  long id = (long)blockIdx.x * 256 + threadIdx.x;
  if (id >= (long)N * 2) return;
  long n = id >> 1; int hd = (int)(id & 1);
  float s = 0.0f, d2 = 0.0f;
#pragma unroll
  for (int c2 = 0; c2 < 32; c2++) {
    float v = hh[n * 64 + hd * 32 + c2];
    s  += v * asrc[hd * 32 + c2];
    d2 += v * adst[hd * 32 + c2];
  }
  ah_s[id] = s; ah_d[id] = d2;
}

__global__ void weff_kernel(const float* __restrict__ geW, const float* __restrict__ aedge,
                            float* __restrict__ weff) {
  int tid = threadIdx.x; if (tid >= 128) return;
  int j = tid >> 1, hd = tid & 1;
  float s = 0.0f;
#pragma unroll
  for (int c2 = 0; c2 < 32; c2++) s += geW[j * 64 + hd * 32 + c2] * aedge[hd * 32 + c2];
  weff[j * 2 + hd] = s;
}

__global__ void ae_kernel(const uint16_t* __restrict__ ea_ext, const float* __restrict__ weff,
                          float* __restrict__ ae, int EXT) {
  long i = (long)blockIdx.x * 256 + threadIdx.x;
  if (i >= EXT) return;
  float s0 = 0.0f, s1 = 0.0f;
#pragma unroll 8
  for (int j = 0; j < 64; j++) {
    float v = b2f(ea_ext[i * 64 + j]);
    s0 += v * weff[j * 2];
    s1 += v * weff[j * 2 + 1];
  }
  ae[i * 2] = s0; ae[i * 2 + 1] = s1;
}

__device__ __forceinline__ u32 enc_f32(float a) {
  u32 b = __float_as_uint(a);
  return b ^ ((b & 0x80000000u) ? 0xFFFFFFFFu : 0x80000000u);
}
__device__ __forceinline__ float dec_f32(u32 k) {
  u32 b = (k & 0x80000000u) ? (k ^ 0x80000000u) : ~k;
  return __uint_as_float(b);
}

__global__ void a_kernel(const int* __restrict__ src, const int* __restrict__ dst,
                         const float* __restrict__ ah_s, const float* __restrict__ ah_d,
                         const float* __restrict__ ae, float* __restrict__ abuf,
                         u32* __restrict__ amax, int E, int EXT) {
  long id = (long)blockIdx.x * 256 + threadIdx.x;
  if (id >= (long)EXT * 2) return;
  long i = id >> 1; int hd = (int)(id & 1);
  int es = (i < E) ? src[i] : (int)(i - E);
  int ed = (i < E) ? dst[i] : (int)(i - E);
  float a = ah_s[(long)es * 2 + hd] + ah_d[(long)ed * 2 + hd] + ae[id];
  a = (a >= 0.0f) ? a : 0.2f * a;
  abuf[id] = a;
  atomicMax(&amax[(long)ed * 2 + hd], enc_f32(a));
}

__global__ void ex_kernel(const int* __restrict__ dst, float* __restrict__ abuf,
                          const u32* __restrict__ amax, float* __restrict__ ssum, int E, int EXT) {
  long id = (long)blockIdx.x * 256 + threadIdx.x;
  if (id >= (long)EXT * 2) return;
  long i = id >> 1; int hd = (int)(id & 1);
  int ed = (i < E) ? dst[i] : (int)(i - E);
  float m = dec_f32(amax[(long)ed * 2 + hd]);
  float e = __expf(abuf[id] - m);
  abuf[id] = e;
  atomicAdd(&ssum[(long)ed * 2 + hd], e);
}

__global__ void agg_kernel(const int* __restrict__ src, const int* __restrict__ dst,
                           const float* __restrict__ abuf, const float* __restrict__ ssum,
                           const float* __restrict__ hh, float* __restrict__ hB, int E, int EXT) {
  long id = (long)blockIdx.x * 256 + threadIdx.x;
  if (id >= (long)EXT * 64) return;
  long i = id >> 6; int c2 = (int)(id & 63); int hd = c2 >> 5;
  int es = (i < E) ? src[i] : (int)(i - E);
  int ed = (i < E) ? dst[i] : (int)(i - E);
  float alpha = abuf[i * 2 + hd] / (ssum[(long)ed * 2 + hd] + 1e-16f);
  atomicAdd(&hB[(long)ed * 64 + c2], hh[(long)es * 64 + c2] * alpha);
}

__global__ void gat_post_kernel(const float* __restrict__ hB, const float* __restrict__ gb,
                                const float* __restrict__ lng, const float* __restrict__ lnb,
                                int do_ln, float* __restrict__ hf, uint16_t* __restrict__ hb, int N) {
  int n = blockIdx.x * 256 + threadIdx.x; if (n >= N) return;
  float v[64]; float mu = 0.0f;
#pragma unroll
  for (int d = 0; d < 64; d++) { v[d] = hB[(long)n * 64 + d] + gb[d]; mu += v[d]; }
  if (do_ln) {
    mu *= (1.0f / 64.0f);
    float var = 0.0f;
#pragma unroll
    for (int d = 0; d < 64; d++) { float t = v[d] - mu; var += t * t; }
    var *= (1.0f / 64.0f);
    float inv = rsqrtf(var + 1e-5f);
#pragma unroll
    for (int d = 0; d < 64; d++) {
      float t = (v[d] - mu) * inv * lng[d] + lnb[d];
      t = fmaxf(t, 0.0f);
      hf[(long)n * 64 + d] = t; hb[(long)n * 64 + d] = f2b(t);
    }
  } else {
#pragma unroll
    for (int d = 0; d < 64; d++) { hf[(long)n * 64 + d] = v[d]; hb[(long)n * 64 + d] = f2b(v[d]); }
  }
}

// ---------------------------------------------------------------------------
// Per-graph 32-node self-attention + LayerNorm. One workgroup per graph.
// ---------------------------------------------------------------------------
__global__ __launch_bounds__(256) void attn_kernel(
    const uint16_t* __restrict__ hbin, const float* __restrict__ hfin,
    const uint16_t* __restrict__ Wkt, const float* __restrict__ bk,
    const uint16_t* __restrict__ Wqt, const float* __restrict__ bq,
    const uint16_t* __restrict__ Wvt, const float* __restrict__ bv,
    const float* __restrict__ lng, const float* __restrict__ lnb,
    float* __restrict__ xn) {
  __shared__ __align__(16) uint16_t sx[32 * 64];
  __shared__ __align__(16) uint16_t sk[32 * 64];
  __shared__ __align__(16) uint16_t sq[32 * 64];
  __shared__ __align__(16) uint16_t svT[64 * 32];   // v transposed [d][m]
  __shared__ __align__(16) uint16_t skqb[32 * 32];
  __shared__ float skq[32 * 32];
  __shared__ float sz[32 * 64];
  int gph = blockIdx.x, tid = threadIdx.x;
  for (int idx = tid; idx < 2048; idx += 256) sx[idx] = hbin[(long)gph * 2048 + idx];
  __syncthreads();
  int wid = tid >> 5, lane = tid & 31, g = lane >> 4, r = lane & 15;
  // K/Q/V projections: 24 tiles (3 matrices x 8), 3 per wave
  for (int t = wid; t < 24; t += 8) {
    int mat = t >> 3, tt = t & 7, tm = tt >> 2, tn = tt & 3;
    const uint16_t* Wt = (mat == 0) ? Wkt : ((mat == 1) ? Wqt : Wvt);
    const float* bb = (mat == 0) ? bk : ((mat == 1) ? bq : bv);
    f32x8 acc = {};
    for (int kk = 0; kk < 64; kk += 32) {
      FragBF fa, fb;
      loadA(fa, sx + (tm * 16 + r) * 64 + kk + 8 * g);
      loadB(fb, Wt + (tn * 16 + r) * 64 + kk + 8 * g);
      acc = wmma_bf16(fa, fb, acc);
    }
    int n = tn * 16 + r;
    float bvv = bb[n];
#pragma unroll
    for (int i = 0; i < 8; i++) {
      int m = tm * 16 + i + 8 * g;
      uint16_t h16 = f2b(acc[i] + bvv);
      if (mat == 0)      sk[m * 64 + n] = h16;
      else if (mat == 1) sq[m * 64 + n] = h16;
      else               svT[n * 32 + m] = h16;
    }
  }
  __syncthreads();
  // kq = (k @ q^T) / 8 : 4 tiles
  if (wid < 4) {
    int tm = wid >> 1, tn = wid & 1;
    f32x8 acc = {};
    for (int kk = 0; kk < 64; kk += 32) {
      FragBF fa, fb;
      loadA(fa, sk + (tm * 16 + r) * 64 + kk + 8 * g);
      loadB(fb, sq + (tn * 16 + r) * 64 + kk + 8 * g);
      acc = wmma_bf16(fa, fb, acc);
    }
#pragma unroll
    for (int i = 0; i < 8; i++)
      skq[(tm * 16 + i + 8 * g) * 32 + tn * 16 + r] = acc[i] * 0.125f;
  }
  __syncthreads();
  if (tid < 32) {
    float mx = -1e30f;
    for (int j = 0; j < 32; j++) mx = fmaxf(mx, skq[tid * 32 + j]);
    float s = 0.0f;
    for (int j = 0; j < 32; j++) s += __expf(skq[tid * 32 + j] - mx);
    float invs = 1.0f / s;
    for (int j = 0; j < 32; j++) skqb[tid * 32 + j] = f2b(__expf(skq[tid * 32 + j] - mx) * invs);
  }
  __syncthreads();
  // z = kq @ v : 8 tiles, K=32 (one step)
  {
    int tm = wid >> 2, tn = wid & 3;
    f32x8 acc = {};
    FragBF fa, fb;
    loadA(fa, skqb + (tm * 16 + r) * 32 + 8 * g);
    loadB(fb, svT + (tn * 16 + r) * 32 + 8 * g);
    acc = wmma_bf16(fa, fb, acc);
#pragma unroll
    for (int i = 0; i < 8; i++)
      sz[(tm * 16 + i + 8 * g) * 64 + tn * 16 + r] = acc[i];
  }
  __syncthreads();
  if (tid < 32) {
    long node = (long)gph * 32 + tid;
    float zr[64]; float mu = 0.0f;
#pragma unroll
    for (int d = 0; d < 64; d++) { zr[d] = hfin[node * 64 + d] + sz[tid * 64 + d]; mu += zr[d]; }
    mu *= (1.0f / 64.0f);
    float var = 0.0f;
#pragma unroll
    for (int d = 0; d < 64; d++) { float t = zr[d] - mu; var += t * t; }
    var *= (1.0f / 64.0f);
    float inv = rsqrtf(var + 1e-5f);
#pragma unroll
    for (int d = 0; d < 64; d++)
      xn[node * 64 + d] = (zr[d] - mu) * inv * lng[d] + lnb[d];
  }
}

// ---------------------------------------------------------------------------
// Final edge MLP kernels
// ---------------------------------------------------------------------------
__global__ void xv_kernel(const float* __restrict__ xn, const float* __restrict__ ea,
                          const int* __restrict__ src, const int* __restrict__ dst,
                          uint16_t* __restrict__ xv, int E) {
  long id = (long)blockIdx.x * 256 + threadIdx.x;
  if (id >= (long)E * 128) return;
  long e = id >> 7; int c2 = (int)(id & 127);
  float v;
  if (c2 < 64) {
    int s0 = src[e], d0 = dst[e];
    v = fmaxf(xn[(long)s0 * 64 + c2], xn[(long)d0 * 64 + c2]);
  } else {
    v = ea[e * 64 + (c2 - 64)];
  }
  xv[id] = f2b(v);
}

__global__ void eb_kernel(const uint16_t* __restrict__ t2, const float* __restrict__ W2,
                          const float* __restrict__ b2, const int* __restrict__ src,
                          const int* __restrict__ batch, float* __restrict__ out, int E) {
  int e = blockIdx.x * 256 + threadIdx.x; if (e >= E) return;
  float s = b2[0];
#pragma unroll 8
  for (int j = 0; j < 128; j++) s += b2f(t2[(long)e * 128 + j]) * W2[j];
  s = (s >= 0.0f) ? s : 0.01f * s;
  atomicAdd(&out[batch[src[e]]], s);
}

// ---------------------------------------------------------------------------
// Host side
// ---------------------------------------------------------------------------
static void run_gemm(hipStream_t stream, const uint16_t* A, const uint16_t* Bt, const float* bias,
                     float* Cf, uint16_t* Cb, int M, int N, int K, int act, bool outb, bool accum) {
  long tiles = (long)(M >> 5) * (N >> 6);      // 32x64 macro-tiles per wave
  dim3 grid((unsigned)((tiles + 7) / 8)), block(256);
  if (accum) {
    gemm_bf16_kernel<ACT_NONE, false, true><<<grid, block, 0, stream>>>(A, Bt, bias, Cf, Cb, M, N, K);
  } else if (outb) {
    if (act == ACT_NONE)
      gemm_bf16_kernel<ACT_NONE, true, false><<<grid, block, 0, stream>>>(A, Bt, bias, Cf, Cb, M, N, K);
    else if (act == ACT_RELU)
      gemm_bf16_kernel<ACT_RELU, true, false><<<grid, block, 0, stream>>>(A, Bt, bias, Cf, Cb, M, N, K);
    else
      gemm_bf16_kernel<ACT_LRELU, true, false><<<grid, block, 0, stream>>>(A, Bt, bias, Cf, Cb, M, N, K);
  } else {
    if (act == ACT_NONE)
      gemm_bf16_kernel<ACT_NONE, false, false><<<grid, block, 0, stream>>>(A, Bt, bias, Cf, Cb, M, N, K);
    else
      gemm_bf16_kernel<ACT_LRELU, false, false><<<grid, block, 0, stream>>>(A, Bt, bias, Cf, Cb, M, N, K);
  }
}

#define KL(kern, n, ...) kern<<<dim3((unsigned)(((long)(n) + 255) / 256)), dim3(256), 0, stream>>>(__VA_ARGS__)

extern "C" void kernel_launch(void* const* d_in, const int* in_sizes, int n_in,
                              void* d_out, int out_size, void* d_ws, size_t ws_size,
                              hipStream_t stream) {
  (void)in_sizes; (void)n_in; (void)out_size; (void)ws_size;
  const int G = 1024, Ecnt = 131072, Nn = 32768;
  const int EXT = Ecnt + Nn;

  const int*   x        = (const int*)d_in[0];
  const int*   src      = (const int*)d_in[1];
  const int*   dst      = src + Ecnt;
  const int*   edge_attr= (const int*)d_in[2];
  const int*   batch    = (const int*)d_in[3];
  const float* c        = (const float*)d_in[4];
  const float* elem_e   = (const float*)d_in[5];
  const float* type_emb = (const float*)d_in[6];
  const float* posi_W   = (const float*)d_in[7];
  const float* posi_b   = (const float*)d_in[8];
  const float* npn_W    = (const float*)d_in[9];
  const float* npn_b    = (const float*)d_in[10];
  const float* pred_W1  = (const float*)d_in[11];
  const float* pred_b1  = (const float*)d_in[12];
  const float* pred_W2  = (const float*)d_in[13];
  const float* pred_b2  = (const float*)d_in[14];
  const float* pred_W3  = (const float*)d_in[15];
  const float* pred_b3  = (const float*)d_in[16];
  const float* node_W   = (const float*)d_in[17];
  const float* node_b   = (const float*)d_in[18];
  const float* edge_emb = (const float*)d_in[19];
  const float* em_W1    = (const float*)d_in[20];
  const float* em_b1    = (const float*)d_in[21];
  const float* em_W2    = (const float*)d_in[22];
  const float* em_b2    = (const float*)d_in[23];
  const float* gat_W    = (const float*)d_in[24];
  const float* gat_eW   = (const float*)d_in[25];
  const float* gat_b    = (const float*)d_in[26];
  const float* att_src  = (const float*)d_in[27];
  const float* att_dst  = (const float*)d_in[28];
  const float* att_edge = (const float*)d_in[29];
  const float* gat_ln_g = (const float*)d_in[30];
  const float* gat_ln_b = (const float*)d_in[31];
  const float* att_Wk   = (const float*)d_in[32];
  const float* att_bk   = (const float*)d_in[33];
  const float* att_Wq   = (const float*)d_in[34];
  const float* att_bq   = (const float*)d_in[35];
  const float* att_Wv   = (const float*)d_in[36];
  const float* att_bv   = (const float*)d_in[37];
  const float* att_ln_g = (const float*)d_in[38];
  const float* att_ln_b = (const float*)d_in[39];
  const float* lin_W    = (const float*)d_in[40];
  const float* lin_b    = (const float*)d_in[41];
  const float* e2f_W1   = (const float*)d_in[42];
  const float* e2f_b1   = (const float*)d_in[43];
  const float* e2f_W2   = (const float*)d_in[44];
  const float* e2f_b2   = (const float*)d_in[45];
  const float* ef_W1    = (const float*)d_in[46];
  const float* ef_b1    = (const float*)d_in[47];
  const float* ef_W2    = (const float*)d_in[48];
  const float* ef_b2    = (const float*)d_in[49];
  float* out = (float*)d_out;

  // -------- workspace bump allocator --------
  char* ws = (char*)d_ws;
  size_t off = 0;
  auto alloc = [&](size_t bytes) -> char* {
    char* p = ws + off;
    off = (off + bytes + 255) & ~(size_t)255;
    return p;
  };
  // bf16 transposed weights
  uint16_t* w_npn   = (uint16_t*)alloc((size_t)64 * 192 * 2);
  uint16_t* w_pred1 = (uint16_t*)alloc((size_t)64 * 4096 * 2);
  uint16_t* w_pred2 = (uint16_t*)alloc((size_t)64 * 64 * 2);
  uint16_t* w_em1   = (uint16_t*)alloc((size_t)64 * 64 * 2);
  uint16_t* w_em2   = (uint16_t*)alloc((size_t)64 * 64 * 2);
  uint16_t* w_gat   = (uint16_t*)alloc((size_t)6 * 64 * 64 * 2);
  uint16_t* w_k     = (uint16_t*)alloc((size_t)64 * 64 * 2);
  uint16_t* w_q     = (uint16_t*)alloc((size_t)64 * 64 * 2);
  uint16_t* w_v     = (uint16_t*)alloc((size_t)64 * 64 * 2);
  uint16_t* w_lin   = (uint16_t*)alloc((size_t)128 * 128 * 2);
  uint16_t* w_e2f1  = (uint16_t*)alloc((size_t)128 * 128 * 2);
  uint16_t* w_e2f2  = (uint16_t*)alloc((size_t)128 * 128 * 2);
  uint16_t* w_ef1   = (uint16_t*)alloc((size_t)128 * 128 * 2);
  // activations / scratch
  float* ps2t  = (float*)alloc((size_t)Ecnt * 3 * 4);
  float* weff  = (float*)alloc((size_t)64 * 2 * 4);
  float* ah_s  = (float*)alloc((size_t)Nn * 2 * 4);
  float* ah_d  = (float*)alloc((size_t)Nn * 2 * 4);
  float* ae    = (float*)alloc((size_t)EXT * 2 * 4);
  float* abuf  = (float*)alloc((size_t)EXT * 2 * 4);
  u32*   amax  = (u32*)alloc((size_t)Nn * 2 * 4);
  float* ssum  = (float*)alloc((size_t)Nn * 2 * 4);
  float* cnt   = (float*)alloc((size_t)Nn * 4);
  float* ea_f  = (float*)alloc((size_t)Ecnt * 64 * 4);
  uint16_t* ea_ext = (uint16_t*)alloc((size_t)EXT * 64 * 2);
  float* loopb = (float*)alloc((size_t)Nn * 64 * 4);
  float* hf    = (float*)alloc((size_t)Nn * 64 * 4);
  uint16_t* hb = (uint16_t*)alloc((size_t)Nn * 64 * 2);
  float* hh    = (float*)alloc((size_t)Nn * 64 * 4);
  float* hB    = (float*)alloc((size_t)Nn * 64 * 4);
  float* xn    = (float*)alloc((size_t)Nn * 64 * 4);
  float* h2    = (float*)alloc((size_t)G * 64 * 4);
  uint16_t* h1 = (uint16_t*)alloc((size_t)G * 64 * 2);
  // overlaid big regions (phase-exclusive)
  char* bigA = alloc((size_t)Ecnt * 128 * 4);  // vnin (E*192 bf16) | Cbuf (E*128 f32)
  char* bigB = alloc((size_t)Ecnt * 128 * 2);  // vn (E*64 bf16) | em tmps | xv (E*128 bf16)
  char* bigC = alloc((size_t)Ecnt * 128 * 2);  // feat (G*4096 bf16) | t1/t2 (E*128 bf16)
  uint16_t* vnin = (uint16_t*)bigA;
  float*    Cbuf = (float*)bigA;
  uint16_t* vn   = (uint16_t*)bigB;
  uint16_t* xv   = (uint16_t*)bigB;
  uint16_t* emt0 = (uint16_t*)bigB;
  uint16_t* emt1 = (uint16_t*)(bigB + (size_t)Ecnt * 64 * 2);
  uint16_t* feat = (uint16_t*)bigC;
  uint16_t* t1   = (uint16_t*)bigC;

  // -------- weight conversion / transposition --------
  KL(transpose_to_bf16, 64 * 192, npn_W, w_npn, 192, 64);
  KL(transpose_to_bf16, 64 * 4096, pred_W1, w_pred1, 4096, 64);
  KL(transpose_to_bf16, 64 * 64, pred_W2, w_pred2, 64, 64);
  KL(transpose_to_bf16, 64 * 64, em_W1, w_em1, 64, 64);
  KL(transpose_to_bf16, 64 * 64, em_W2, w_em2, 64, 64);
  for (int l = 0; l < 6; l++)
    KL(transpose_to_bf16, 64 * 64, gat_W + (size_t)l * 4096, w_gat + (size_t)l * 4096, 64, 64);
  KL(transpose_to_bf16, 64 * 64, att_Wk, w_k, 64, 64);
  KL(transpose_to_bf16, 64 * 64, att_Wq, w_q, 64, 64);
  KL(transpose_to_bf16, 64 * 64, att_Wv, w_v, 64, 64);
  KL(transpose_to_bf16, 128 * 128, lin_W, w_lin, 128, 128);
  KL(transpose_to_bf16, 128 * 128, e2f_W1, w_e2f1, 128, 128);
  KL(transpose_to_bf16, 128 * 128, e2f_W2, w_e2f2, 128, 128);
  KL(transpose_to_bf16, 128 * 128, ef_W1, w_ef1, 128, 128);

  KL(fill_f32_kernel, G, out, 0.0f, (long)G);

  // -------- Phase A: geometry branch -> Ec --------
  KL(edge_geom_kernel, Ecnt, c, src, dst, ps2t, Ecnt);
  KL(vnin_kernel, (long)Ecnt * 64, ps2t, src, dst, type_emb, x, posi_W, posi_b, vnin, Ecnt);
  run_gemm(stream, vnin, w_npn, npn_b, nullptr, vn, Ecnt, 64, 192, ACT_NONE, true, false);
  feat_kernel<<<dim3(G), dim3(256), 0, stream>>>(ps2t, vn, feat);
  run_gemm(stream, feat, w_pred1, pred_b1, nullptr, h1, G, 64, 4096, ACT_LRELU, true, false);
  run_gemm(stream, h1, w_pred2, pred_b2, h2, nullptr, G, 64, 64, ACT_LRELU, false, false);
  KL(ec_kernel, G, h2, pred_W3, pred_b3, out, G);
  KL(ea_energy_kernel, Nn, x, batch, elem_e, out, Nn);

  // -------- node features + edge embeddings --------
  KL(xf_kernel, (long)Nn * 64, x, node_W, node_b, hf, hb, Nn);
  KL(ea0_kernel, (long)Ecnt * 64, edge_attr, edge_emb, emt0, Ecnt);
  run_gemm(stream, emt0, w_em1, em_b1, nullptr, emt1, Ecnt, 64, 64, ACT_RELU, true, false);
  run_gemm(stream, emt1, w_em2, em_b2, ea_f, nullptr, Ecnt, 64, 64, ACT_NONE, false, false);
  KL(f32_to_bf16_kernel, (long)Ecnt * 64, ea_f, ea_ext, (long)Ecnt * 64);
  KL(fill_f32_kernel, Nn, cnt, 0.0f, (long)Nn);
  KL(cnt_kernel, Ecnt, dst, cnt, Ecnt);
  KL(fill_f32_kernel, (long)Nn * 64, loopb, 0.0f, (long)Nn * 64);
  KL(loop_acc_kernel, (long)Ecnt * 64, dst, ea_f, loopb, Ecnt);
  KL(loop_fin_kernel, (long)Nn * 64, loopb, cnt, ea_ext + (size_t)Ecnt * 64, Nn);

  // -------- GAT layers --------
  for (int l = 0; l < 6; l++) {
    run_gemm(stream, hb, w_gat + (size_t)l * 4096, nullptr, hh, nullptr, Nn, 64, 64, ACT_NONE, false, false);
    KL(ah_kernel, (long)Nn * 2, hh, att_src + (size_t)l * 64, att_dst + (size_t)l * 64, ah_s, ah_d, Nn);
    weff_kernel<<<dim3(1), dim3(128), 0, stream>>>(gat_eW + (size_t)l * 4096, att_edge + (size_t)l * 64, weff);
    KL(ae_kernel, EXT, ea_ext, weff, ae, EXT);
    KL(fill_u32_kernel, (long)Nn * 2, amax, 0x007FFFFFu, (long)Nn * 2);
    KL(fill_f32_kernel, (long)Nn * 2, ssum, 0.0f, (long)Nn * 2);
    KL(fill_f32_kernel, (long)Nn * 64, hB, 0.0f, (long)Nn * 64);
    KL(a_kernel, (long)EXT * 2, src, dst, ah_s, ah_d, ae, abuf, amax, Ecnt, EXT);
    KL(ex_kernel, (long)EXT * 2, dst, abuf, amax, ssum, Ecnt, EXT);
    KL(agg_kernel, (long)EXT * 64, src, dst, abuf, ssum, hh, hB, Ecnt, EXT);
    KL(gat_post_kernel, Nn, hB, gat_b + (size_t)l * 64, gat_ln_g + (size_t)l * 64,
       gat_ln_b + (size_t)l * 64, (l < 5) ? 1 : 0, hf, hb, Nn);
  }

  // -------- per-graph self-attention + LN --------
  attn_kernel<<<dim3(G), dim3(256), 0, stream>>>(hb, hf, w_k, att_bk, w_q, att_bq,
                                                 w_v, att_bv, att_ln_g, att_ln_b, xn);

  // -------- final edge MLPs -> Eb --------
  KL(xv_kernel, (long)Ecnt * 128, xn, ea_f, src, dst, xv, Ecnt);
  run_gemm(stream, xv, w_e2f1, e2f_b1, nullptr, t1, Ecnt, 128, 128, ACT_RELU, true, false);
  run_gemm(stream, xv, w_lin, lin_b, Cbuf, nullptr, Ecnt, 128, 128, ACT_NONE, false, false);
  run_gemm(stream, t1, w_e2f2, e2f_b2, Cbuf, nullptr, Ecnt, 128, 128, ACT_NONE, false, true);
  KL(f32_to_bf16_kernel, (long)Ecnt * 128, Cbuf, xv, (long)Ecnt * 128);
  run_gemm(stream, xv, w_ef1, ef_b1, nullptr, t1, Ecnt, 128, 128, ACT_RELU, true, false);
  KL(eb_kernel, Ecnt, t1, ef_W2, ef_b2, src, batch, out, Ecnt);
}
